// MMdual_attention_23699629539655
// MI455X (gfx1250) — matrix operand order repaired and checked by reading, not compiled, not run on hardware
//
#include <hip/hip_runtime.h>
#include <hip/hip_bf16.h>

typedef __attribute__((ext_vector_type(16))) __bf16 v16bf;
typedef __attribute__((ext_vector_type(8)))  float  v8f;

// ---------- helpers ----------
__device__ __forceinline__ unsigned short f2bf(float f) {
    unsigned int x = __builtin_bit_cast(unsigned int, f);
    unsigned int r = x + 0x7fffu + ((x >> 16) & 1u);   // round-to-nearest-even
    return (unsigned short)(r >> 16);
}
__device__ __forceinline__ float bf2f(unsigned short u) {
    unsigned int x = ((unsigned int)u) << 16;
    return __builtin_bit_cast(float, x);
}
__device__ __forceinline__ __bf16 us2bf(unsigned short u) {
    return __builtin_bit_cast(__bf16, u);
}

// A-fragment (16x32 bf16) fill: row = base + ln, K offsets per ISA layout
__device__ __forceinline__ v16bf load_afrag(const unsigned short* __restrict__ row,
                                            int k0, int hi) {
    v16bf a;
#pragma unroll
    for (int h = 0; h < 16; ++h)
        a[h] = us2bf(row[k0 + (h < 8 ? h : h + 8) + hi * 8]);
    return a;
}
// B-fragment (32x16 bf16) fill from "column-rows" contiguous in K
__device__ __forceinline__ v16bf load_bfrag(const unsigned short* __restrict__ row,
                                            int k0, int hi) {
    v16bf b;
#pragma unroll
    for (int h = 0; h < 16; ++h)
        b[h] = us2bf(row[k0 + (h < 8 ? h : h + 8) + hi * 8]);
    return b;
}

// ---------- convert f32 -> bf16 ----------
__global__ void k_f32_to_bf16(const float* __restrict__ src,
                              unsigned short* __restrict__ dst, int n) {
    int i = blockIdx.x * blockDim.x + threadIdx.x;
    if (i < n) dst[i] = f2bf(src[i]);
}

// ---------- bf16 WMMA GEMM, 64x32 tile per wave (4x2 wmma): C = A * Bw^T ----------
__global__ void __launch_bounds__(256)
__attribute__((amdgpu_waves_per_eu(1)))
k_gemm_bf16(const unsigned short* __restrict__ A,
            const unsigned short* __restrict__ Bw,
            float* __restrict__ C,
            int M, int N, int K) {
    int wave = blockIdx.x * (blockDim.x >> 5) + (threadIdx.x >> 5);
    int ntn  = N >> 5;                          // 32-wide col tiles
    int mt   = wave / ntn;
    int nt   = wave - mt * ntn;
    if (mt >= (M >> 6)) return;                 // wave-uniform
    int lane = threadIdx.x & 31;
    int hi   = lane >> 4;
    int ln   = lane & 15;

    const unsigned short* arow[4];
    const unsigned short* brow[2];
#pragma unroll
    for (int i = 0; i < 4; ++i)
        arow[i] = A + (size_t)(mt * 64 + i * 16 + ln) * K;
#pragma unroll
    for (int j = 0; j < 2; ++j)
        brow[j] = Bw + (size_t)(nt * 32 + j * 16 + ln) * K;

    v8f acc[4][2];
#pragma unroll
    for (int i = 0; i < 4; ++i)
#pragma unroll
        for (int j = 0; j < 2; ++j) acc[i][j] = (v8f){};

    for (int k0 = 0; k0 < K; k0 += 32) {
        v16bf af[4], bfr[2];
#pragma unroll
        for (int i = 0; i < 4; ++i) af[i]  = load_afrag(arow[i], k0, hi);
#pragma unroll
        for (int j = 0; j < 2; ++j) bfr[j] = load_bfrag(brow[j], k0, hi);
#pragma unroll
        for (int i = 0; i < 4; ++i)
#pragma unroll
            for (int j = 0; j < 2; ++j)
                acc[i][j] = __builtin_amdgcn_wmma_f32_16x16x32_bf16(
                    false, af[i], false, bfr[j], (short)0, acc[i][j], false, false);
    }
#pragma unroll
    for (int i = 0; i < 4; ++i)
#pragma unroll
        for (int r = 0; r < 8; ++r) {
            float* crow = C + (size_t)(mt * 64 + i * 16 + r + 8 * hi) * N + nt * 32;
#pragma unroll
            for (int j = 0; j < 2; ++j)
                crow[j * 16 + ln] = acc[i][j][r];
        }
}

// ---------- QKV epilogue: per-head RMSNorm + RoPE, layout for attention ----------
// qkv: [B*Nst, 3C] f32.  Outputs: qb/kb [B,H,Ntot,D] bf16, vt [B,H,D,Ntot] bf16.
__global__ void k_qkv_epilogue(const float* __restrict__ qkv,
                               const float* __restrict__ qw,
                               const float* __restrict__ kw,
                               const float* __restrict__ cs,
                               const float* __restrict__ sn,
                               unsigned short* __restrict__ qb,
                               unsigned short* __restrict__ kb,
                               unsigned short* __restrict__ vt,
                               int Nst, int n_off) {
    const int H = 16, D = 64, C = 1024, Ntot = 2048;
    int wave = blockIdx.x * (blockDim.x >> 5) + (threadIdx.x >> 5);
    int h  = wave & (H - 1);
    int bn = wave >> 4;                 // b*Nst + n
    int n  = bn % Nst;
    int lane = threadIdx.x & 31;
    int d0 = 2 * lane, d1 = 2 * lane + 1;

    const float* base = qkv + (size_t)bn * (3 * C);
    float q0 = base[h * D + d0],         q1 = base[h * D + d1];
    float k0 = base[C + h * D + d0],     k1 = base[C + h * D + d1];
    float v0 = base[2 * C + h * D + d0], v1 = base[2 * C + h * D + d1];

    float sq = q0 * q0 + q1 * q1;
    float sk = k0 * k0 + k1 * k1;
#pragma unroll
    for (int m = 16; m; m >>= 1) {
        sq += __shfl_xor(sq, m, 32);
        sk += __shfl_xor(sk, m, 32);
    }
    float rq = rsqrtf(sq * (1.0f / 64.0f) + 1e-5f);
    float rk = rsqrtf(sk * (1.0f / 64.0f) + 1e-5f);
    q0 *= rq * qw[d0]; q1 *= rq * qw[d1];
    k0 *= rk * kw[d0]; k1 *= rk * kw[d1];

    float c0 = cs[n * D + d0], s0 = sn[n * D + d0];
    float c1 = cs[n * D + d1], s1 = sn[n * D + d1];
    float qe = q0 * c0 - q1 * s0, qo = q1 * c1 + q0 * s1;   // interleaved-pair RoPE
    float ke = k0 * c0 - k1 * s0, ko = k1 * c1 + k0 * s1;

    int b = bn / Nst;
    size_t qk_base = ((size_t)(b * H + h) * Ntot + (n_off + n)) * D;
    qb[qk_base + d0] = f2bf(qe); qb[qk_base + d1] = f2bf(qo);
    kb[qk_base + d0] = f2bf(ke); kb[qk_base + d1] = f2bf(ko);
    size_t vt_base = (size_t)(b * H + h) * D * Ntot + (n_off + n);
    vt[vt_base + (size_t)d0 * Ntot] = f2bf(v0);
    vt[vt_base + (size_t)d1 * Ntot] = f2bf(v1);
}

// ---------- flash attention: 1 wave per (bh, 32-row q block = 2 wmma q-tiles) ----------
__global__ void __launch_bounds__(256)
__attribute__((amdgpu_waves_per_eu(1)))
k_attention(const unsigned short* __restrict__ qb,
            const unsigned short* __restrict__ kb,
            const unsigned short* __restrict__ vt,
            unsigned short* __restrict__ ob) {
    const int D = 64, Ntot = 2048, QB = 2;
    __shared__ unsigned short P[8][QB][16 * 32];
    int wv   = threadIdx.x >> 5;
    int wave = blockIdx.x * 8 + wv;
    int qt = wave & 63;             // Ntot/32 = 64 q-blocks
    int bh = wave >> 6;             // 0..31
    int lane = threadIdx.x & 31;
    int hi = lane >> 4, ln = lane & 15;

    // Q A-fragments, pre-scaled by 1/sqrt(D)=0.125 (exact exponent shift)
    v16bf aq[QB][2];
#pragma unroll
    for (int g = 0; g < QB; ++g) {
        const unsigned short* qrow =
            qb + ((size_t)bh * Ntot + qt * 32 + g * 16 + ln) * D;
#pragma unroll
        for (int c = 0; c < 2; ++c)
#pragma unroll
            for (int h = 0; h < 16; ++h) {
                int kk = c * 32 + (h < 8 ? h : h + 8) + hi * 8;
                aq[g][c][h] = us2bf(f2bf(bf2f(qrow[kk]) * 0.125f));
            }
    }

    float mrow[QB][8], lrow[QB][8];
    v8f o[QB][4];
#pragma unroll
    for (int g = 0; g < QB; ++g) {
#pragma unroll
        for (int r = 0; r < 8; ++r) { mrow[g][r] = -1e30f; lrow[g][r] = 0.0f; }
#pragma unroll
        for (int dt = 0; dt < 4; ++dt) o[g][dt] = (v8f){};
    }

    const unsigned short* kbbase = kb + (size_t)bh * Ntot * D;
    const unsigned short* vtbase = vt + (size_t)bh * D * Ntot;

    for (int kt = 0; kt < Ntot; kt += 32) {
        if (kt + 32 < Ntot) {   // wave-uniform branch; prefetch next chunk
            __builtin_prefetch(kbbase + (size_t)(kt + 32 + ln) * D, 0, 1);
            __builtin_prefetch(vtbase + (size_t)ln * Ntot + kt + 32, 0, 1);
        }
        // scores for both q sub-tiles: shared K B-fragments
        v8f s[QB][2];
#pragma unroll
        for (int g = 0; g < QB; ++g) { s[g][0] = (v8f){}; s[g][1] = (v8f){}; }
#pragma unroll
        for (int t = 0; t < 2; ++t) {
            const unsigned short* krow = kbbase + (size_t)(kt + t * 16 + ln) * D;
#pragma unroll
            for (int c = 0; c < 2; ++c) {
                v16bf bk = load_bfrag(krow, c * 32, hi);
#pragma unroll
                for (int g = 0; g < QB; ++g)
                    s[g][t] = __builtin_amdgcn_wmma_f32_16x16x32_bf16(
                        false, aq[g][c], false, bk, (short)0, s[g][t], false, false);
            }
        }
        // online softmax per q sub-tile (row r+8*hi; masks<16 stay within half-wave)
#pragma unroll
        for (int g = 0; g < QB; ++g) {
            float mnew[8], alpha[8];
#pragma unroll
            for (int r = 0; r < 8; ++r) {
                float v = fmaxf(s[g][0][r], s[g][1][r]);
                for (int msk = 8; msk; msk >>= 1) v = fmaxf(v, __shfl_xor(v, msk, 32));
                mnew[r]  = fmaxf(mrow[g][r], v);
                alpha[r] = __expf(mrow[g][r] - mnew[r]);
            }
#pragma unroll
            for (int r = 0; r < 8; ++r) {
                float p0 = __expf(s[g][0][r] - mnew[r]);
                float p1 = __expf(s[g][1][r] - mnew[r]);
                float rs = p0 + p1;
                for (int msk = 8; msk; msk >>= 1) rs += __shfl_xor(rs, msk, 32);
                lrow[g][r] = lrow[g][r] * alpha[r] + rs;
                mrow[g][r] = mnew[r];
#pragma unroll
                for (int dt = 0; dt < 4; ++dt) o[g][dt][r] *= alpha[r];
                P[wv][g][(r + 8 * hi) * 32 + ln]      = f2bf(p0);
                P[wv][g][(r + 8 * hi) * 32 + 16 + ln] = f2bf(p1);
            }
        }
        // reload P as A fragments (same-wave LDS is in-order)
        v16bf ap[QB];
#pragma unroll
        for (int g = 0; g < QB; ++g)
#pragma unroll
            for (int h = 0; h < 16; ++h) {
                int kk = (h < 8 ? h : h + 8) + hi * 8;
                ap[g][h] = us2bf(P[wv][g][ln * 32 + kk]);
            }
        // P·V with shared V B-fragments
#pragma unroll
        for (int dt = 0; dt < 4; ++dt) {
            const unsigned short* vrow = vtbase + (size_t)(dt * 16 + ln) * Ntot + kt;
            v16bf bv = load_bfrag(vrow, 0, hi);
#pragma unroll
            for (int g = 0; g < QB; ++g)
                o[g][dt] = __builtin_amdgcn_wmma_f32_16x16x32_bf16(
                    false, ap[g], false, bv, (short)0, o[g][dt], false, false);
        }
    }
    // normalize + store bf16 [B, Ntot, C] rows for the projection GEMM
    int b = bh >> 4, h = bh & 15;
#pragma unroll
    for (int g = 0; g < QB; ++g)
#pragma unroll
        for (int dt = 0; dt < 4; ++dt)
#pragma unroll
            for (int r = 0; r < 8; ++r) {
                int q = qt * 32 + g * 16 + r + 8 * hi;
                float val = o[g][dt][r] / lrow[g][r];
                ob[(size_t)(b * Ntot + q) * 1024 + h * 64 + dt * 16 + ln] = f2bf(val);
            }
}

// ---------- output projection, 64x32 per wave, per-stream routing + bias ----------
__global__ void __launch_bounds__(256)
__attribute__((amdgpu_waves_per_eu(1)))
k_proj(const unsigned short* __restrict__ A,   // [4096, 1024]
       const unsigned short* __restrict__ Wx,
       const unsigned short* __restrict__ Wa,
       const float* __restrict__ bx,
       const float* __restrict__ ba,
       float* __restrict__ out) {
    const int K = 1024, N = 1024, N1 = 1536, N2 = 512, Ntot = 2048;
    int wave = blockIdx.x * (blockDim.x >> 5) + (threadIdx.x >> 5);
    int nt = wave & 31;                    // N/32 = 32
    int mt = wave >> 5;                    // 0..63
    int lane = threadIdx.x & 31;
    int hi = lane >> 4, ln = lane & 15;

    int m0 = mt * 64;
    int b  = m0 / Ntot;
    int n  = m0 % Ntot;
    bool is_xs = (n < N1);                 // 64-row tile never straddles boundary
    const unsigned short* W    = is_xs ? Wx : Wa;
    const float*          bias = is_xs ? bx : ba;

    const unsigned short* arow[4];
    const unsigned short* brow[2];
#pragma unroll
    for (int i = 0; i < 4; ++i)
        arow[i] = A + (size_t)(m0 + i * 16 + ln) * K;
#pragma unroll
    for (int j = 0; j < 2; ++j)
        brow[j] = W + (size_t)(nt * 32 + j * 16 + ln) * K;

    v8f acc[4][2];
#pragma unroll
    for (int i = 0; i < 4; ++i)
#pragma unroll
        for (int j = 0; j < 2; ++j) acc[i][j] = (v8f){};

    for (int k0 = 0; k0 < K; k0 += 32) {
        v16bf af[4], bfr[2];
#pragma unroll
        for (int i = 0; i < 4; ++i) af[i]  = load_afrag(arow[i], k0, hi);
#pragma unroll
        for (int j = 0; j < 2; ++j) bfr[j] = load_bfrag(brow[j], k0, hi);
#pragma unroll
        for (int i = 0; i < 4; ++i)
#pragma unroll
            for (int j = 0; j < 2; ++j)
                acc[i][j] = __builtin_amdgcn_wmma_f32_16x16x32_bf16(
                    false, af[i], false, bfr[j], (short)0, acc[i][j], false, false);
    }
    float bval[2];
#pragma unroll
    for (int j = 0; j < 2; ++j) bval[j] = bias[nt * 32 + j * 16 + ln];
#pragma unroll
    for (int i = 0; i < 4; ++i)
#pragma unroll
        for (int r = 0; r < 8; ++r) {
            int rr = i * 16 + r + 8 * hi;      // row within 64-row tile
            float* orow = is_xs
                ? out + (size_t)(b * N1 + n + rr) * N
                : out + (size_t)2 * N1 * N + (size_t)(b * N2 + (n - N1) + rr) * N;
#pragma unroll
            for (int j = 0; j < 2; ++j)
                orow[nt * 32 + j * 16 + ln] = acc[i][j][r] + bval[j];
        }
}

// ---------- host launcher ----------
extern "C" void kernel_launch(void* const* d_in, const int* in_sizes, int n_in,
                              void* d_out, int out_size, void* d_ws, size_t ws_size,
                              hipStream_t stream) {
    const int B = 2, N1 = 1536, N2 = 512, C = 1024, Ntot = 2048, H = 16, D = 64;

    const float* x     = (const float*)d_in[1];
    const float* y     = (const float*)d_in[2];
    const float* cos1  = (const float*)d_in[3];
    const float* sin1  = (const float*)d_in[4];
    const float* cos2  = (const float*)d_in[5];
    const float* sin2  = (const float*)d_in[6];
    const float* Wqx   = (const float*)d_in[7];
    const float* Wqa   = (const float*)d_in[8];
    const float* xsqw  = (const float*)d_in[9];
    const float* xskw  = (const float*)d_in[10];
    const float* auqw  = (const float*)d_in[11];
    const float* aukw  = (const float*)d_in[12];
    const float* Wpx   = (const float*)d_in[13];
    const float* bpx   = (const float*)d_in[14];
    const float* Wpa   = (const float*)d_in[15];
    const float* bpa   = (const float*)d_in[16];
    float* out = (float*)d_out;

    char* ws = (char*)d_ws;
    size_t off = 0;
    auto alloc = [&](size_t bytes) -> void* {
        void* p = ws + off;
        off = (off + bytes + 255) & ~(size_t)255;
        return p;
    };
    unsigned short* xb   = (unsigned short*)alloc((size_t)B * N1 * C * 2);
    unsigned short* yb   = (unsigned short*)alloc((size_t)B * N2 * C * 2);
    unsigned short* Wqxb = (unsigned short*)alloc((size_t)3 * C * C * 2);
    unsigned short* Wqab = (unsigned short*)alloc((size_t)3 * C * C * 2);
    unsigned short* Wpxb = (unsigned short*)alloc((size_t)C * C * 2);
    unsigned short* Wpab = (unsigned short*)alloc((size_t)C * C * 2);
    float* qkv_xs        = (float*)alloc((size_t)B * N1 * 3 * C * 4);
    float* qkv_au        = (float*)alloc((size_t)B * N2 * 3 * C * 4);
    unsigned short* qb   = (unsigned short*)alloc((size_t)B * H * Ntot * D * 2);
    unsigned short* kb   = (unsigned short*)alloc((size_t)B * H * Ntot * D * 2);
    unsigned short* vt   = (unsigned short*)alloc((size_t)B * H * D * Ntot * 2);
    unsigned short* ob   = (unsigned short*)alloc((size_t)B * Ntot * C * 2);

    // 1) conversions to bf16
    auto conv = [&](const float* s, unsigned short* d, int n) {
        k_f32_to_bf16<<<(n + 255) / 256, 256, 0, stream>>>(s, d, n);
    };
    conv(x,   xb,   B * N1 * C);
    conv(y,   yb,   B * N2 * C);
    conv(Wqx, Wqxb, 3 * C * C);
    conv(Wqa, Wqab, 3 * C * C);
    conv(Wpx, Wpxb, C * C);
    conv(Wpa, Wpab, C * C);

    // 2) QKV GEMMs   (M x 3C) = (M x C) * (3C x C)^T   (64x32 tiles per wave)
    {
        int M = B * N1, N = 3 * C, tiles = (M / 64) * (N / 32);   // 48*96
        k_gemm_bf16<<<tiles / 8, 256, 0, stream>>>(xb, Wqxb, qkv_xs, M, N, C);
    }
    {
        int M = B * N2, N = 3 * C, tiles = (M / 64) * (N / 32);   // 16*96
        k_gemm_bf16<<<tiles / 8, 256, 0, stream>>>(yb, Wqab, qkv_au, M, N, C);
    }

    // 3) RMSNorm + RoPE + layout
    {
        int waves = B * N1 * H;
        k_qkv_epilogue<<<waves / 8, 256, 0, stream>>>(qkv_xs, xsqw, xskw, cos1, sin1,
                                                      qb, kb, vt, N1, 0);
    }
    {
        int waves = B * N2 * H;
        k_qkv_epilogue<<<waves / 8, 256, 0, stream>>>(qkv_au, auqw, aukw, cos2, sin2,
                                                      qb, kb, vt, N2, N1);
    }

    // 4) joint flash attention: 32 bh * 64 q-blocks = 2048 waves
    k_attention<<<256, 256, 0, stream>>>(qb, kb, vt, ob);

    // 5) output projections (xs then au, concatenated in d_out)
    {
        int tiles = (B * Ntot / 64) * (C / 32);   // 64 * 32
        k_proj<<<tiles / 8, 256, 0, stream>>>(ob, Wpxb, Wpab, bpx, bpa, out);
    }
}
